// MultiHeadAttention_77309411992
// MI455X (gfx1250) — matrix operand order repaired and checked by reading, not compiled, not run on hardware
//
#include <hip/hip_runtime.h>
#include <hip/hip_bf16.h>

// ---------------------------------------------------------------------------
// MHA on gfx1250 (MI455X): f16 WMMA (fp32 accumulate) flash-attention pipeline
// with TDM (tensor_load_to_lds) staging of K/V tiles.
// B=4, S=2048, D=1024, H=16, DK=64.
// ---------------------------------------------------------------------------

typedef __attribute__((ext_vector_type(16))) _Float16 v16h;
typedef __attribute__((ext_vector_type(8)))  _Float16 v8h;
typedef __attribute__((ext_vector_type(8)))  float    v8f;
typedef __attribute__((ext_vector_type(4)))  float    v4f;
typedef __attribute__((ext_vector_type(4)))  _Float16 v4h;
typedef __attribute__((ext_vector_type(4)))  unsigned int tdm_v4u;
typedef __attribute__((ext_vector_type(8)))  int          tdm_v8i;
typedef __attribute__((ext_vector_type(4)))  int          tdm_v4i;

#define WMMA_F16(A, Bm, C) \
  __builtin_amdgcn_wmma_f32_16x16x32_f16(false, (A), false, (Bm), (short)0, (C), false, false)

#define HAS_TDM __has_builtin(__builtin_amdgcn_tensor_load_to_lds)

static constexpr int Bb  = 4;
static constexpr int Ss  = 2048;
static constexpr int Dd  = 1024;
static constexpr int Hh  = 16;
static constexpr int DK  = 64;
static constexpr int BS  = Bb * Ss;          // 8192 rows
static constexpr int LP  = 72;               // padded LDS row stride (halves)

// ---------------------------------------------------------------------------
// fp32 -> f16 conversion (4 elements / thread, vectorized)
// ---------------------------------------------------------------------------
__global__ __launch_bounds__(256) void cvt_f32_f16(const float* __restrict__ in,
                                                   _Float16* __restrict__ out,
                                                   size_t n4) {
  size_t i = (size_t)blockIdx.x * blockDim.x + threadIdx.x;
  if (i >= n4) return;
  v4f x = *(const v4f*)(in + i * 4);
  v4h y;
  y.x = (_Float16)x.x; y.y = (_Float16)x.y; y.z = (_Float16)x.z; y.w = (_Float16)x.w;
  *(v4h*)(out + i * 4) = y;
}

// ---------------------------------------------------------------------------
// WMMA GEMM: Y[m,n] = sum_k X[m,k] * W[n,k] + bias[n]   (torch Linear)
//   Block 256 thr = 8 waves; block tile 128x128; wave tile 32x64.
//   mode 0: f16 out, [B,H,S,DK] layout, *scale      (Q / K)
//   mode 1: f16 out, [B,H,DK,S] transposed layout   (V)
//   mode 2: f32 out, [M,N] row-major (+bias)        (final projection)
// ---------------------------------------------------------------------------
__global__ __launch_bounds__(256)
void proj_gemm(const _Float16* __restrict__ X, const _Float16* __restrict__ W,
               const float* __restrict__ bias, void* __restrict__ out,
               float scale, int mode) {
  const int KD = Dd;
  const int m_blk = blockIdx.x * 128;
  const int n_blk = blockIdx.y * 128;
  const int wave  = threadIdx.x >> 5;
  const int lane  = threadIdx.x & 31;
  const int wm = wave & 3, wn = wave >> 2;
  const int m0 = m_blk + wm * 32;
  const int n0 = n_blk + wn * 64;
  const int laneHi = lane >> 4, lane15 = lane & 15;
  const int b0 = laneHi * 8;          // A-fragment K sub-offset per lane half
  const int bK = laneHi * 16;         // unused (B fragments also use b0)

  v8f acc[2][4] = {};

  const _Float16* xr0 = X + (size_t)(m0 + lane15) * KD;        // A rows, mi=0
  const _Float16* xr1 = X + (size_t)(m0 + 16 + lane15) * KD;   // A rows, mi=1
  const _Float16* wr  = W + (size_t)(n0 + lane15) * KD;        // B col n = W row n

  for (int k0 = 0; k0 < KD; k0 += 32) {
    v16h a0, a1;
    *(v8h*)&a0       = *(const v8h*)(xr0 + k0 + b0);
    *((v8h*)&a0 + 1) = *(const v8h*)(xr0 + k0 + 16 + b0);
    *(v8h*)&a1       = *(const v8h*)(xr1 + k0 + b0);
    *((v8h*)&a1 + 1) = *(const v8h*)(xr1 + k0 + 16 + b0);
#pragma unroll
    for (int ni = 0; ni < 4; ++ni) {
      const _Float16* p = wr + (size_t)(ni * 16) * KD + k0 + b0;
      v16h bf;
      *(v8h*)&bf       = *(const v8h*)(p);
      *((v8h*)&bf + 1) = *(const v8h*)(p + 16);
      acc[0][ni] = WMMA_F16(a0, bf, acc[0][ni]);
      acc[1][ni] = WMMA_F16(a1, bf, acc[1][ni]);
    }
  }

#pragma unroll
  for (int mi = 0; mi < 2; ++mi) {
#pragma unroll
    for (int ni = 0; ni < 4; ++ni) {
      const int n = n0 + ni * 16 + lane15;
      const float bv = bias[n];
#pragma unroll
      for (int r = 0; r < 8; ++r) {
        const int m = m0 + mi * 16 + laneHi * 8 + r;
        const float v = (acc[mi][ni][r] + bv) * scale;
        if (mode == 0) {                 // [B,H,S,DK] f16
          const int b = m >> 11, s = m & (Ss - 1);
          const int h = n >> 6,  dk = n & (DK - 1);
          ((_Float16*)out)[(((size_t)b * Hh + h) * Ss + s) * DK + dk] = (_Float16)v;
        } else if (mode == 1) {          // [B,H,DK,S] f16 (V transposed)
          const int b = m >> 11, s = m & (Ss - 1);
          const int h = n >> 6,  dk = n & (DK - 1);
          ((_Float16*)out)[(((size_t)b * Hh + h) * DK + dk) * Ss + s] = (_Float16)v;
        } else {                         // f32 [M, D]
          ((float*)out)[(size_t)m * Dd + n] = v;
        }
      }
    }
  }
}

// ---------------------------------------------------------------------------
// TDM: issue a 2-D tensor_load_to_lds of a [tile_d1 rows x tile_d0 elems]
// f16 tile, with LDS padding of 16B after every 128B (row stride 144B = 72
// halves).  Descriptor packing per CDNA5 ISA 8.3/8.4 (D# groups 0/1).
// This toolchain exposes the 6-arg builtin:
//   (uint32x4 g0, int32x8 g1, int32x4 g2, int32x4 g3, int32x8 g4, i32 cpol)
// ---------------------------------------------------------------------------
#if HAS_TDM
__device__ __forceinline__ void tdm_load_tile_f16(unsigned lds_byte_addr,
                                                  const void* gaddr,
                                                  unsigned tensor_d0,
                                                  unsigned tensor_d1,
                                                  unsigned tile_d0,
                                                  unsigned tile_d1,
                                                  unsigned long long d0_stride) {
  const unsigned long long ga = (unsigned long long)(uintptr_t)gaddr;
  tdm_v4u g0;
  g0.x = 1u;                                            // count=1 (valid), user
  g0.y = lds_byte_addr;                                 // lds_addr[31:0]
  g0.z = (unsigned)(ga & 0xFFFFFFFFu);                  // global_addr[31:0]
  g0.w = (unsigned)((ga >> 32) & 0x01FFFFFFu)           // global_addr[56:32]
       | (2u << 30);                                    // type = 2 ("image")
  tdm_v8i g1;
  g1[0] = (int)((1u << 16)        // data_size = 1 -> 2 bytes
              | (1u << 20)        // pad_enable
              | (4u << 22)        // pad_interval: 4 -> every 32 DWORDs (128B)
              | (3u << 25));      // pad_amount:   3 -> 4 DWORDs (16B)
  g1[1] = (int)((tensor_d0 & 0xFFFFu) << 16);                       // dim0[15:0]
  g1[2] = (int)(((tensor_d0 >> 16) & 0xFFFFu)
              | ((tensor_d1 & 0xFFFFu) << 16));                     // dim0 hi | dim1 lo
  g1[3] = (int)(((tensor_d1 >> 16) & 0xFFFFu)
              | ((tile_d0 & 0xFFFFu) << 16));                       // dim1 hi | tile0
  g1[4] = (int)(tile_d1 & 0xFFFFu);                                 // tile1 | tile2=0
  g1[5] = (int)(d0_stride & 0xFFFFFFFFu);                           // dim0_stride lo
  g1[6] = (int)((d0_stride >> 32) & 0xFFFFu);                       // stride hi | dim1_stride=0
  g1[7] = 0;
  const tdm_v4i z4 = {0, 0, 0, 0};                                  // 2-D: groups 2/3 unused
  const tdm_v8i z8 = {0, 0, 0, 0, 0, 0, 0, 0};
  __builtin_amdgcn_tensor_load_to_lds(g0, g1, z4, z4, z8, 0);
}
#endif

// ---------------------------------------------------------------------------
// Flash attention: grid (B*H, S/64), block 128 (4 waves x 16 q-rows).
//   Q pre-scaled by 1/sqrt(DK); causal => only kb <= qb blocks visited.
//   K/V 64x64 tiles staged in padded LDS via TDM (or cooperative copy),
//   shared by all 4 waves.  Online softmax in fp32; P round-trips through a
//   per-wave padded LDS tile to convert C-layout -> A-layout.
// ---------------------------------------------------------------------------
__global__ __launch_bounds__(128)
void flash_attn(const _Float16* __restrict__ Q, const _Float16* __restrict__ K,
                const _Float16* __restrict__ Vt, _Float16* __restrict__ O) {
  const int bh = blockIdx.x;                 // b*H + h
  const int qb = blockIdx.y * 64;            // q block base
  const int wave = threadIdx.x >> 5;
  const int lane = threadIdx.x & 31;
  const int laneHi = lane >> 4, lane15 = lane & 15;
  const int b0 = laneHi * 8;
  const int bK = laneHi * 16;
  const int qw = qb + wave * 16;             // this wave's q rows [qw, qw+16)

  // LDS: K tile (64 x 72), V tile (64 x 72), P tiles (4 waves x 16 x 72)
  __shared__ __align__(16) _Float16 smem[64 * LP + 64 * LP + 4 * 16 * LP];
  _Float16* Ksh = smem;
  _Float16* Vsh = smem + 64 * LP;
  _Float16* Psh = smem + 2 * 64 * LP + wave * 16 * LP;

  const _Float16* Kg = K  + (size_t)bh * Ss * DK;   // [S, DK] for this head
  const _Float16* Vg = Vt + (size_t)bh * DK * Ss;   // [DK, S] for this head

  // Q A-fragments (row = qw + lane15), kept in registers for all K blocks.
  v16h qa[2];
  const _Float16* qrow = Q + ((size_t)bh * Ss + qw + lane15) * DK;
#pragma unroll
  for (int ks = 0; ks < 2; ++ks) {
    *(v8h*)&qa[ks]       = *(const v8h*)(qrow + ks * 32 + b0);
    *((v8h*)&qa[ks] + 1) = *(const v8h*)(qrow + ks * 32 + 16 + b0);
  }

  float mrun[8], lrun[8];
#pragma unroll
  for (int r = 0; r < 8; ++r) { mrun[r] = -1e30f; lrun[r] = 0.f; }
  v8f oacc[4] = {};

  for (int kb = 0; kb <= qb; kb += 64) {
    // ---- stage K[kb..kb+63, :] and Vt[:, kb..kb+63] into padded LDS ----
#if HAS_TDM
    if (wave == 0) {
      tdm_load_tile_f16((unsigned)(uintptr_t)Ksh, Kg + (size_t)kb * DK,
                        /*tensor_d0=*/DK, /*tensor_d1=*/(unsigned)(Ss - kb),
                        /*tile_d0=*/DK, /*tile_d1=*/64, /*d0_stride=*/DK);
      tdm_load_tile_f16((unsigned)(uintptr_t)Vsh, Vg + kb,
                        /*tensor_d0=*/Ss, /*tensor_d1=*/DK,
                        /*tile_d0=*/64, /*tile_d1=*/DK, /*d0_stride=*/Ss);
      __builtin_amdgcn_s_wait_tensorcnt(0);
    }
#else
    // cooperative fallback: 64 rows x 8 chunks of 8 halves, padded stride
    for (int i = threadIdx.x; i < 64 * 8; i += 128) {
      const int r = i >> 3, c = (i & 7) * 8;
      *(v8h*)&Ksh[r * LP + c] = *(const v8h*)(Kg + (size_t)(kb + r) * DK + c);
      *(v8h*)&Vsh[r * LP + c] = *(const v8h*)(Vg + (size_t)r * Ss + kb + c);
    }
#endif
    __syncthreads();   // publish tiles (uniform: kb loop bound is block-uniform)

    // ---- scores S = Q * K^T  (64 keys, 4 N-tiles) ----
    v8f sc[4] = {};
#pragma unroll
    for (int ks = 0; ks < 2; ++ks) {
#pragma unroll
      for (int ni = 0; ni < 4; ++ni) {
        const _Float16* p = &Ksh[(ni * 16 + lane15) * LP + ks * 32 + b0];
        v16h bf;
        *(v8h*)&bf       = *(const v8h*)(p);
        *((v8h*)&bf + 1) = *(const v8h*)(p + 16);
        sc[ni] = WMMA_F16(qa[ks], bf, sc[ni]);
      }
    }

    // ---- causal mask (diagonal block only) ----
    if (kb == qb) {
#pragma unroll
      for (int ni = 0; ni < 4; ++ni) {
        const int kg = kb + ni * 16 + lane15;
#pragma unroll
        for (int r = 0; r < 8; ++r) {
          const int qg = qw + laneHi * 8 + r;
          if (kg > qg) sc[ni][r] = -1e9f;
        }
      }
    }

    // ---- online softmax (row stats live per 16-lane half) ----
    float pfac[8];
#pragma unroll
    for (int r = 0; r < 8; ++r) {
      float mx = fmaxf(fmaxf(sc[0][r], sc[1][r]), fmaxf(sc[2][r], sc[3][r]));
      mx = fmaxf(mx, __shfl_xor(mx, 1));
      mx = fmaxf(mx, __shfl_xor(mx, 2));
      mx = fmaxf(mx, __shfl_xor(mx, 4));
      mx = fmaxf(mx, __shfl_xor(mx, 8));
      const float mnew = fmaxf(mrun[r], mx);
      const float corr = __expf(mrun[r] - mnew);
      float rs = 0.f;
#pragma unroll
      for (int ni = 0; ni < 4; ++ni) {
        const float p = __expf(sc[ni][r] - mnew);
        sc[ni][r] = p;
        rs += p;
      }
      rs += __shfl_xor(rs, 1);
      rs += __shfl_xor(rs, 2);
      rs += __shfl_xor(rs, 4);
      rs += __shfl_xor(rs, 8);
      lrun[r] = lrun[r] * corr + rs;
      mrun[r] = mnew;
      pfac[r] = corr;
    }
#pragma unroll
    for (int ni = 0; ni < 4; ++ni)
#pragma unroll
      for (int r = 0; r < 8; ++r) oacc[ni][r] *= pfac[r];

    // ---- P: C-layout -> LDS (f16) -> A-layout fragments ----
#pragma unroll
    for (int ni = 0; ni < 4; ++ni)
#pragma unroll
      for (int r = 0; r < 8; ++r)
        Psh[(laneHi * 8 + r) * LP + ni * 16 + lane15] = (_Float16)sc[ni][r];
    __syncthreads();

    // ---- O += P * V  (V fragments from the staged LDS tile) ----
#pragma unroll
    for (int ks = 0; ks < 2; ++ks) {
      const _Float16* pp = &Psh[lane15 * LP + ks * 32 + b0];
      v16h pa;
      *(v8h*)&pa       = *(const v8h*)(pp);
      *((v8h*)&pa + 1) = *(const v8h*)(pp + 16);
#pragma unroll
      for (int ni = 0; ni < 4; ++ni) {
        const _Float16* vp = &Vsh[(ni * 16 + lane15) * LP + ks * 32 + b0];
        v16h vf;
        *(v8h*)&vf       = *(const v8h*)(vp);
        *((v8h*)&vf + 1) = *(const v8h*)(vp + 16);
        oacc[ni] = WMMA_F16(pa, vf, oacc[ni]);
      }
    }
    __syncthreads();   // protect K/V/P tiles before next iteration's staging
  }

  // ---- finalize O / l, store to [B,S,D] f16 ----
  const int b = bh >> 4, h = bh & 15;
#pragma unroll
  for (int ni = 0; ni < 4; ++ni) {
    const int dk = ni * 16 + lane15;
#pragma unroll
    for (int r = 0; r < 8; ++r) {
      const int s = qw + laneHi * 8 + r;
      const float v = oacc[ni][r] / lrun[r];
      O[((size_t)b * Ss + s) * Dd + h * DK + dk] = (_Float16)v;
    }
  }
}

// ---------------------------------------------------------------------------
// Host-side launch
// ---------------------------------------------------------------------------
extern "C" void kernel_launch(void* const* d_in, const int* in_sizes, int n_in,
                              void* d_out, int out_size, void* d_ws, size_t ws_size,
                              hipStream_t stream) {
  const float* q  = (const float*)d_in[0];
  const float* k  = (const float*)d_in[1];
  const float* v  = (const float*)d_in[2];
  // d_in[3] = causal mask (implicit in flash_attn)
  const float* Wq = (const float*)d_in[4];  const float* bq = (const float*)d_in[5];
  const float* Wk = (const float*)d_in[6];  const float* bk = (const float*)d_in[7];
  const float* Wv = (const float*)d_in[8];  const float* bv = (const float*)d_in[9];
  const float* Wo = (const float*)d_in[10]; const float* bo = (const float*)d_in[11];

  const size_t ACT = (size_t)BS * Dd;            // 8388608 elems
  const size_t WSZ = (size_t)Dd * Dd;            // 1048576 elems
  char* ws = (char*)d_ws;
  _Float16* Xq  = (_Float16*)(ws);
  _Float16* Xk  = Xq + ACT;
  _Float16* Xv  = Xk + ACT;
  _Float16* Wq6 = Xv + ACT;
  _Float16* Wk6 = Wq6 + WSZ;
  _Float16* Wv6 = Wk6 + WSZ;
  _Float16* Wo6 = Wv6 + WSZ;
  _Float16* Qh  = Wo6 + WSZ;                     // [B,H,S,DK]
  _Float16* Kh  = Qh + ACT;                      // [B,H,S,DK]
  _Float16* Vt  = Kh + ACT;                      // [B,H,DK,S]
  _Float16* Oh  = Vt + ACT;                      // [B,S,D]

  {
    const int thr = 256;
    const int gA = (int)(ACT / 4 / thr);
    const int gW = (int)(WSZ / 4 / thr);
    cvt_f32_f16<<<gA, thr, 0, stream>>>(q, Xq, ACT / 4);
    cvt_f32_f16<<<gA, thr, 0, stream>>>(k, Xk, ACT / 4);
    cvt_f32_f16<<<gA, thr, 0, stream>>>(v, Xv, ACT / 4);
    cvt_f32_f16<<<gW, thr, 0, stream>>>(Wq, Wq6, WSZ / 4);
    cvt_f32_f16<<<gW, thr, 0, stream>>>(Wk, Wk6, WSZ / 4);
    cvt_f32_f16<<<gW, thr, 0, stream>>>(Wv, Wv6, WSZ / 4);
    cvt_f32_f16<<<gW, thr, 0, stream>>>(Wo, Wo6, WSZ / 4);
  }

  dim3 pg(BS / 128, Dd / 128);
  proj_gemm<<<pg, 256, 0, stream>>>(Xq, Wq6, bq, (void*)Qh, 0.125f, 0);
  proj_gemm<<<pg, 256, 0, stream>>>(Xk, Wk6, bk, (void*)Kh, 1.0f,   0);
  proj_gemm<<<pg, 256, 0, stream>>>(Xv, Wv6, bv, (void*)Vt, 1.0f,   1);

  dim3 ag(Bb * Hh, Ss / 64);
  flash_attn<<<ag, 128, 0, stream>>>(Qh, Kh, Vt, Oh);

  proj_gemm<<<pg, 256, 0, stream>>>(Oh, Wo6, bo, d_out, 1.0f, 2);
}
